// ScaledDotProductAttention_549755814004
// MI455X (gfx1250) — matrix operand order repaired and checked by reading, not compiled, not run on hardware
//
#include <hip/hip_runtime.h>

// ---------------------------------------------------------------------------
// ScaledDotProductAttention for MI455X (gfx1250, wave32, WMMA + async LDS)
//   q,k,v: [8,2048,64] f32.  Outputs (concat): out[8,2048,64],
//   attn[8,2048,2048], log_attn[8,2048,2048], all f32.
// Since q,k are L2-normalized and divided by 25, scores are in [-0.04,0.04];
// exp never overflows, so lse = log(sum exp(s)) needs no max subtraction.
// K tiles (and Q) are staged with GLOBAL_LOAD_ASYNC_TO_LDS_B128 into
// double-buffered LDS; V is register-staged and scatter-transposed.
// ---------------------------------------------------------------------------

typedef __attribute__((ext_vector_type(16))) _Float16 v16h;
typedef __attribute__((ext_vector_type(8)))  _Float16 h8;
typedef __attribute__((ext_vector_type(2)))  _Float16 h2;
typedef __attribute__((ext_vector_type(8)))  float    v8f;

union Frag { v16h v; h8 h[2]; };

constexpr int Bn   = 8;     // batch
constexpr int L    = 2048;  // seq len (q and k)
constexpr int D    = 64;    // head dim (q/k and v)
constexpr int QB   = 64;    // query rows per workgroup (4 waves x 16)
constexpr int KT   = 64;    // key tile staged in LDS
constexpr int NT   = L / KT;
constexpr int LSTR = 72;    // padded LDS row stride in halves (144B)
constexpr int ASTR = 40;    // padded stride for per-wave attn tile (16x32)

__device__ __forceinline__ v8f wmma_f16(const Frag& a, const Frag& b, v8f c) {
  return __builtin_amdgcn_wmma_f32_16x16x32_f16(false, a.v, false, b.v,
                                                (short)0, c, false, false);
}

// 16B per-lane async copy global -> LDS (tracked by ASYNCcnt)
__device__ __forceinline__ void async_copy16(const void* gptr, void* lptr) {
  uint32_t laddr = (uint32_t)(uintptr_t)lptr;   // low 32 bits == LDS address
  asm volatile("global_load_async_to_lds_b128 %0, %1, off"
               :: "v"(laddr), "v"((uint64_t)(uintptr_t)gptr)
               : "memory");
}

__device__ __forceinline__ void wait_async0() {
#if __has_builtin(__builtin_amdgcn_s_wait_asynccnt)
  __builtin_amdgcn_s_wait_asynccnt(0);
#else
  asm volatile("s_wait_asynccnt 0x0" ::: "memory");
#endif
}

// stage a KT x D (or QB x D) f16 tile: 512 16B chunks over 128 threads
__device__ __forceinline__ void stage_async(const _Float16* __restrict__ g,
                                            _Float16* l, int tid) {
  #pragma unroll
  for (int i = 0; i < 4; ++i) {
    int idx = tid + i * 128;
    int r = idx >> 3, c = (idx & 7) * 8;
    async_copy16(g + (size_t)r * D + c, l + r * LSTR + c);
  }
}

// ---------------------------------------------------------------------------
// Kernel 1: normalize + convert to f16.  One wave per 64-elem row.
//   qn = q/(25*max(|q|,eps)); kn = k/max(|k|,eps); vh = (f16)v
// ---------------------------------------------------------------------------
__global__ __launch_bounds__(256) void prep_kernel(
    const float* __restrict__ q, const float* __restrict__ k,
    const float* __restrict__ v, _Float16* __restrict__ qn,
    _Float16* __restrict__ kn, _Float16* __restrict__ vh) {
  const int BL   = Bn * L;
  const int wave = threadIdx.x >> 5;
  const int lane = threadIdx.x & 31;
  const int rid  = blockIdx.x * 8 + wave;      // 0 .. 3*BL-1
  const int sec  = rid / BL;
  const int r    = rid - sec * BL;

  const float* src = (sec == 0 ? q : (sec == 1 ? k : v)) + (size_t)r * D;
  float2 x = *(const float2*)(src + lane * 2);

  float scale = 1.0f;
  if (sec < 2) {
    float ss = x.x * x.x + x.y * x.y;
    #pragma unroll
    for (int off = 16; off > 0; off >>= 1) ss += __shfl_xor(ss, off, 32);
    float n = sqrtf(ss);
    scale = 1.0f / fmaxf(n, 1e-12f);
    if (sec == 0) scale *= 0.04f;              // 1/temperature = 1/25
  }

  _Float16* dst = (sec == 0 ? qn : (sec == 1 ? kn : vh)) + (size_t)r * D;
  h2 o;
  o[0] = (_Float16)(x.x * scale);
  o[1] = (_Float16)(x.y * scale);
  *(h2*)(dst + lane * 2) = o;
}

// ---------------------------------------------------------------------------
// Kernel 2: lse[b,i] = log( sum_j exp( qn_i . kn_j ) )
// Double-buffered async K staging; 4 waves x 16 query rows.
// ---------------------------------------------------------------------------
__global__ __launch_bounds__(128) void lse_kernel(
    const _Float16* __restrict__ qn, const _Float16* __restrict__ kn,
    float* __restrict__ lse) {
  __shared__ _Float16 Qs[QB * LSTR];
  __shared__ _Float16 Ks[2][KT * LSTR];

  const int tid = threadIdx.x;
  const int w   = tid >> 5;
  const int lane = tid & 31, lo = lane & 15, hi = lane >> 4;
  const int b  = blockIdx.x >> 5;              // 32 blocks per batch
  const int q0 = (blockIdx.x & 31) * QB;

  const _Float16* qg = qn + ((size_t)b * L + q0) * D;
  const _Float16* kg = kn + (size_t)b * L * D;

  stage_async(qg, Qs, tid);
  wait_async0();
  __syncthreads();

  Frag aq0, aq1;
  {
    const _Float16* ap = &Qs[(w * 16 + lo) * LSTR + 8 * hi];
    aq0.h[0] = *(const h8*)(ap);
    aq0.h[1] = *(const h8*)(ap + 16);
    aq1.h[0] = *(const h8*)(ap + 32);
    aq1.h[1] = *(const h8*)(ap + 48);
  }

  float lacc[8];
  #pragma unroll
  for (int r = 0; r < 8; ++r) lacc[r] = 0.f;

  stage_async(kg, Ks[0], tid);                 // tile 0 in flight

  for (int kt = 0; kt < NT; ++kt) {
    const int cur = kt & 1;
    wait_async0();                             // this wave's K[cur] landed
    __syncthreads();                           // everyone's K[cur]; buf cur^1 free
    if (kt + 1 < NT)
      stage_async(kg + (size_t)(kt + 1) * KT * D, Ks[cur ^ 1], tid);

    #pragma unroll
    for (int ns = 0; ns < 4; ++ns) {
      const _Float16* bp = &Ks[cur][(ns * 16 + lo) * LSTR + 8 * hi];
      Frag b0, b1;
      b0.h[0] = *(const h8*)(bp);      b0.h[1] = *(const h8*)(bp + 16);
      b1.h[0] = *(const h8*)(bp + 32); b1.h[1] = *(const h8*)(bp + 48);
      v8f c;
      #pragma unroll
      for (int r = 0; r < 8; ++r) c[r] = 0.f;
      c = wmma_f16(aq0, b0, c);
      c = wmma_f16(aq1, b1, c);
      #pragma unroll
      for (int r = 0; r < 8; ++r) lacc[r] += __expf(c[r]);
    }
  }

  #pragma unroll
  for (int r = 0; r < 8; ++r) {
    float s = lacc[r];
    #pragma unroll
    for (int off = 1; off < 16; off <<= 1) s += __shfl_xor(s, off, 32);
    if (lo == 0)
      lse[(size_t)b * L + q0 + w * 16 + 8 * hi + r] = __logf(s);
  }
}

// ---------------------------------------------------------------------------
// Kernel 3: recompute scores; emit attn & log_attn (non-temporal streaming
// stores); accumulate out = attn @ V.  Double-buffered: K via async LDS
// copies, V via register staging + scatter transpose.
// ---------------------------------------------------------------------------
__global__ __launch_bounds__(128) void attn_out_kernel(
    const _Float16* __restrict__ qn, const _Float16* __restrict__ kn,
    const _Float16* __restrict__ vh, const float* __restrict__ lse,
    float* __restrict__ outp, float* __restrict__ attn,
    float* __restrict__ lattn) {
  __shared__ _Float16 Qs[QB * LSTR];
  __shared__ _Float16 Ks[2][KT * LSTR];
  __shared__ _Float16 Vt[2][D * LSTR];     // transposed: Vt[buf][dv][key]
  __shared__ _Float16 As[4][16 * ASTR];    // per-wave 16x32 attn tile (f16)

  const int tid = threadIdx.x;
  const int w   = tid >> 5;
  const int lane = tid & 31, lo = lane & 15, hi = lane >> 4;
  const int b  = blockIdx.x >> 5;
  const int q0 = (blockIdx.x & 31) * QB;

  const _Float16* qg = qn + ((size_t)b * L + q0) * D;
  const _Float16* kg = kn + (size_t)b * L * D;
  const _Float16* vg = vh + (size_t)b * L * D;

  stage_async(qg, Qs, tid);
  wait_async0();
  __syncthreads();

  Frag aq0, aq1;
  {
    const _Float16* ap = &Qs[(w * 16 + lo) * LSTR + 8 * hi];
    aq0.h[0] = *(const h8*)(ap);
    aq0.h[1] = *(const h8*)(ap + 16);
    aq1.h[0] = *(const h8*)(ap + 32);
    aq1.h[1] = *(const h8*)(ap + 48);
  }

  float lseq[8];
  #pragma unroll
  for (int r = 0; r < 8; ++r)
    lseq[r] = lse[(size_t)b * L + q0 + w * 16 + 8 * hi + r];

  v8f cacc[4];
  #pragma unroll
  for (int n = 0; n < 4; ++n)
    #pragma unroll
    for (int r = 0; r < 8; ++r) cacc[n][r] = 0.f;

  const size_t rowbase = (size_t)b * L + q0 + w * 16 + 8 * hi;  // + r later
  _Float16* Aw = As[w];

  // V register staging (each thread owns 4 x 16B chunks of the tile)
  h8 vreg[4];
  auto loadV = [&](int kt) {
    #pragma unroll
    for (int i = 0; i < 4; ++i) {
      int idx = tid + i * 128;
      int r = idx >> 3, c = (idx & 7) * 8;
      vreg[i] = *(const h8*)(vg + (size_t)(kt * KT + r) * D + c);
    }
  };
  auto scatterV = [&](int buf) {
    #pragma unroll
    for (int i = 0; i < 4; ++i) {
      int idx = tid + i * 128;
      int r = idx >> 3, c = (idx & 7) * 8;
      #pragma unroll
      for (int e = 0; e < 8; ++e) Vt[buf][(c + e) * LSTR + r] = vreg[i][e];
    }
  };

  stage_async(kg, Ks[0], tid);                 // K tile 0 in flight
  loadV(0);                                    // V tile 0 in registers

  for (int kt = 0; kt < NT; ++kt) {
    const int cur = kt & 1;
    wait_async0();                             // this wave's K[cur] landed
    scatterV(cur);                             // Vt[cur] (buf free per barrier kt-1)
    __syncthreads();                           // all K[cur] + Vt[cur] visible
    if (kt + 1 < NT) {
      stage_async(kg + (size_t)(kt + 1) * KT * D, Ks[cur ^ 1], tid);
      loadV(kt + 1);
    }

    #pragma unroll
    for (int ks = 0; ks < 2; ++ks) {           // two 32-key chunks
      #pragma unroll
      for (int nsub = 0; nsub < 2; ++nsub) {   // two 16-key score tiles
        const int n0 = ks * 32 + nsub * 16;
        const _Float16* bp = &Ks[cur][(n0 + lo) * LSTR + 8 * hi];
        Frag b0, b1;
        b0.h[0] = *(const h8*)(bp);      b0.h[1] = *(const h8*)(bp + 16);
        b1.h[0] = *(const h8*)(bp + 32); b1.h[1] = *(const h8*)(bp + 48);
        v8f c;
        #pragma unroll
        for (int r = 0; r < 8; ++r) c[r] = 0.f;
        c = wmma_f16(aq0, b0, c);
        c = wmma_f16(aq1, b1, c);

        const size_t colg = (size_t)kt * KT + n0 + lo;
        #pragma unroll
        for (int r = 0; r < 8; ++r) {
          float la = c[r] - lseq[r];           // log_softmax
          float at = __expf(la);               // softmax
          size_t off = (rowbase + r) * L + colg;
          __builtin_nontemporal_store(at, attn + off);
          __builtin_nontemporal_store(la, lattn + off);
          Aw[(8 * hi + r) * ASTR + nsub * 16 + lo] = (_Float16)at;
        }
      }
      // A fragment = this wave's 16x32 attn chunk (same-wave LDS ops in order)
      Frag a;
      {
        const _Float16* ap = &Aw[lo * ASTR + 16 * hi];
        a.h[0] = *(const h8*)(ap);
        a.h[1] = *(const h8*)(ap + 8);
      }
      // out += attn_chunk @ V_chunk  (N = 64 in four 16-wide subtiles)
      #pragma unroll
      for (int n = 0; n < 4; ++n) {
        const _Float16* bp = &Vt[cur][(n * 16 + lo) * LSTR + ks * 32 + 16 * hi];
        Frag bv;
        bv.h[0] = *(const h8*)(bp);
        bv.h[1] = *(const h8*)(bp + 8);
        cacc[n] = wmma_f16(a, bv, cacc[n]);
      }
    }
  }

  #pragma unroll
  for (int n = 0; n < 4; ++n)
    #pragma unroll
    for (int r = 0; r < 8; ++r)
      outp[(rowbase + r) * D + n * 16 + lo] = cacc[n][r];
}

// ---------------------------------------------------------------------------
extern "C" void kernel_launch(void* const* d_in, const int* in_sizes, int n_in,
                              void* d_out, int out_size, void* d_ws, size_t ws_size,
                              hipStream_t stream) {
  (void)in_sizes; (void)n_in; (void)out_size; (void)ws_size;
  const float* q = (const float*)d_in[0];
  const float* k = (const float*)d_in[1];
  const float* v = (const float*)d_in[2];

  float* outp  = (float*)d_out;                       // [8,2048,64]
  float* attn  = outp + (size_t)Bn * L * D;           // [8,2048,2048]
  float* lattn = attn + (size_t)Bn * L * L;           // [8,2048,2048]

  // workspace: qn/kn/vh in f16 (2MB each) + lse f32 (64KB)
  _Float16* qn = (_Float16*)d_ws;
  _Float16* kn = qn + (size_t)Bn * L * D;
  _Float16* vh = kn + (size_t)Bn * L * D;
  float*   lse = (float*)(vh + (size_t)Bn * L * D);

  prep_kernel<<<(3 * Bn * L) / 8, 256, 0, stream>>>(q, k, v, qn, kn, vh);
  lse_kernel<<<(Bn * L) / QB, 128, 0, stream>>>(qn, kn, lse);
  attn_out_kernel<<<(Bn * L) / QB, 128, 0, stream>>>(qn, kn, vh, lse,
                                                     outp, attn, lattn);
}
